// Encoder_50611894616749
// MI455X (gfx1250) — matrix operand order repaired and checked by reading, not compiled, not run on hardware
//
#include <hip/hip_runtime.h>

#define DQ 64
#define DD 512
#define SSZ 1024
#define BBZ 4
#define HHZ 8
#define ROWS (BBZ * SSZ) /* 4096 */
#define LLZ 4
#define NSTEP (DD / 32) /* 16 k-steps of 32 */

typedef __attribute__((ext_vector_type(16))) _Float16 v16h;
typedef __attribute__((ext_vector_type(8)))  _Float16 v8h;
typedef __attribute__((ext_vector_type(8)))  float    v8f;
typedef __attribute__((ext_vector_type(4)))  int      v4i;
typedef __attribute__((ext_vector_type(8)))  int      v8i;
typedef __attribute__((ext_vector_type(4)))  unsigned v4u;

#if __has_builtin(__builtin_amdgcn_tensor_load_to_lds)
#define HAVE_TDM 1
#else
#define HAVE_TDM 0
#endif

#if __has_builtin(__builtin_amdgcn_global_load_async_to_lds_b128)
#define HAVE_ASYNC 1
#else
#define HAVE_ASYNC 0
#endif

#if __has_builtin(__builtin_amdgcn_s_wait_asynccnt)
#define WAIT_ASYNC(n) __builtin_amdgcn_s_wait_asynccnt(n)
#else
#define WAIT_ASYNC(n) asm volatile("s_wait_asynccnt %0" ::"i"(n) : "memory")
#endif

#if __has_builtin(__builtin_amdgcn_s_wait_tensorcnt)
#define WAIT_TENSOR(n) __builtin_amdgcn_s_wait_tensorcnt(n)
#else
#define WAIT_TENSOR(n) asm volatile("s_wait_tensorcnt %0" ::"i"(n) : "memory")
#endif

// Build a 16-half A/B fragment from two aligned 8-half chunks.
static __device__ __forceinline__ v16h ld16(const _Float16* p0, const _Float16* p1) {
  v8h lo = *(const v8h*)p0;
  v8h hi = *(const v8h*)p1;
  v16h r;
#pragma unroll
  for (int i = 0; i < 8; ++i) { r[i] = lo[i]; r[i + 8] = hi[i]; }
  return r;
}

static __device__ __forceinline__ v8f wmma_f16(v16h a, v16h b, v8f c) {
  return __builtin_amdgcn_wmma_f32_16x16x32_f16(false, a, false, b, (short)0, c, false, false);
}

#if HAVE_TDM
// TDM descriptor (CDNA5 ISA s8): DMA a 64-row x 32-half tile (row stride
// 512 halves) from global into LDS, hardware-padding 4 dwords after every
// 16 dwords to produce the 40-half LDS row pitch the WMMA readers expect.
static __device__ __forceinline__ void tdm_load_panel(const _Float16* gsrc, unsigned lds_byte_off) {
  unsigned long long ga = (unsigned long long)(size_t)gsrc;
  v4u g0;
  g0[0] = 1u;                                                // count=1, user descriptor
  g0[1] = lds_byte_off;                                      // lds_addr (bytes)
  g0[2] = (unsigned)ga;                                      // global_addr[31:0]
  g0[3] = (unsigned)((ga >> 32) & 0x01FFFFFFu) | (2u << 30); // global_addr[56:32], type=2
  v8i g1;
  g1[0] = (1 << 16)   /* data_size = 2 bytes */
        | (1 << 20)   /* pad_enable */
        | (3 << 22)   /* pad_interval: 16 dwords (one 32-half row) */
        | (3 << 25);  /* pad_amount: 4 dwords -> 40-half pitch */
  g1[1] = (DD & 0xFFFF) << 16;            // tensor_dim0 = 512 (low 16 bits)
  g1[2] = (DD >> 16) | (16384 << 16);     // tensor_dim0 hi, tensor_dim1 = 16384 (low 16)
  g1[3] = 0 | (32 << 16);                 // tensor_dim1 hi, tile_dim0 = 32 elems
  g1[4] = 64;                             // tile_dim1 = 64 rows, tile_dim2 = 0
  g1[5] = DD;                             // tensor_dim0_stride = 512 elems (low 32)
  g1[6] = 0;                              // stride hi, tensor_dim1_stride lo
  g1[7] = 0;
  v4i z4 = {0, 0, 0, 0};
#if defined(__clang_major__) && __clang_major__ >= 23
  v8i z8 = {0, 0, 0, 0, 0, 0, 0, 0};
  __builtin_amdgcn_tensor_load_to_lds(g0, g1, z4, z4, z8, 0);
#else
  __builtin_amdgcn_tensor_load_to_lds(g0, g1, z4, z4, 0);
#endif
}
#endif

// ---------------------------------------------------------------- utilities
__global__ void f32_to_f16_k(const float* __restrict__ in, _Float16* __restrict__ out, int n) {
  int i = blockIdx.x * 256 + threadIdx.x;
  if (i < n) out[i] = (_Float16)in[i];
}

// LayerNorm over D=512; one row per 128-thread block. Writes f16 (out16) or f32 (out32, may alias x).
__global__ void ln_k(const float* __restrict__ x, const float* __restrict__ w,
                     const float* __restrict__ b, _Float16* __restrict__ out16,
                     float* __restrict__ out32) {
  int row = blockIdx.x, t = threadIdx.x;
  const float* xr = x + (size_t)row * DD;
  float v[4]; float s = 0.f, q = 0.f;
#pragma unroll
  for (int i = 0; i < 4; ++i) { v[i] = xr[t + i * 128]; s += v[i]; q += v[i] * v[i]; }
  __shared__ float rs[128], rq[128];
  rs[t] = s; rq[t] = q; __syncthreads();
  for (int off = 64; off; off >>= 1) {
    if (t < off) { rs[t] += rs[t + off]; rq[t] += rq[t + off]; }
    __syncthreads();
  }
  float mu = rs[0] * (1.f / DD);
  float var = rq[0] * (1.f / DD) - mu * mu;
  float rstd = rsqrtf(var + 1e-5f);
#pragma unroll
  for (int i = 0; i < 4; ++i) {
    int c = t + i * 128;
    float y = (v[i] - mu) * rstd * w[c] + b[c];
    if (out16) out16[(size_t)row * DD + c] = (_Float16)y;
    else       out32[(size_t)row * DD + c] = y;
  }
}

// C[M,512] = A[M,512](f16) * Bw[512,512]^T(f16) (+bias) (+resid).
// 4-wave workgroup computes a 64x64 block; 64x32 A/B panels double-buffered in
// LDS. Staging: TDM tensor_load_to_lds (TENSORcnt) if available, else per-lane
// async global->LDS (ASYNCcnt), else sync b128 copies. Overlapped with WMMA.
// Output modes: 0 = f32 [M,N] (+bias+resid); 1 = f16 [B,H,S,DH];
//               2 = f16 [B,H,DH,S] (V^T); 3 = f32 [B,H,S,DH] (R).
__global__ void __launch_bounds__(128)
gemm_k(const _Float16* __restrict__ A, const _Float16* __restrict__ Bw,
       const float* __restrict__ bias, const float* __restrict__ resid,
       float* __restrict__ outf, _Float16* __restrict__ outh, int mode) {
  const int t = threadIdx.x;
  const int wave = t >> 5, lane = t & 31, l16 = lane & 15, hf = (lane >> 4) & 1;
  const int bn = blockIdx.x * 64, bm = blockIdx.y * 64;

  // 40-half row pitch (80B) to spread LDS banks; 16B-aligned sub-chunks.
  __shared__ __align__(16) _Float16 lA[2][64 * 40];
  __shared__ __align__(16) _Float16 lB[2][64 * 40];

  // 256 16B chunks per 64x32 panel; each of the 128 threads moves chunks t, t+128.
  const int r0 = t >> 2,          s0c = t & 3;
  const int r1 = (t + 128) >> 2,  s1c = (t + 128) & 3;
  (void)r0; (void)r1; (void)s0c; (void)s1c;

  auto stage = [&](int buf, int k) {
#if HAVE_TDM
    if (wave == 0) { // one TDM descriptor pair per workgroup per k-step
      tdm_load_panel(A  + (size_t)bm * DD + k, (unsigned)(size_t)(void*)&lA[buf][0]);
      tdm_load_panel(Bw + (size_t)bn * DD + k, (unsigned)(size_t)(void*)&lB[buf][0]);
    }
#elif HAVE_ASYNC
    typedef __attribute__((address_space(1))) v4i* gvp_t;
    typedef __attribute__((address_space(3))) v4i* lvp_t;
    __builtin_amdgcn_global_load_async_to_lds_b128(
        (gvp_t)(const void*)(A + (size_t)(bm + r0) * DD + k + s0c * 8),
        (lvp_t)(void*)&lA[buf][r0 * 40 + s0c * 8], 0, 0);
    __builtin_amdgcn_global_load_async_to_lds_b128(
        (gvp_t)(const void*)(A + (size_t)(bm + r1) * DD + k + s1c * 8),
        (lvp_t)(void*)&lA[buf][r1 * 40 + s1c * 8], 0, 0);
    __builtin_amdgcn_global_load_async_to_lds_b128(
        (gvp_t)(const void*)(Bw + (size_t)(bn + r0) * DD + k + s0c * 8),
        (lvp_t)(void*)&lB[buf][r0 * 40 + s0c * 8], 0, 0);
    __builtin_amdgcn_global_load_async_to_lds_b128(
        (gvp_t)(const void*)(Bw + (size_t)(bn + r1) * DD + k + s1c * 8),
        (lvp_t)(void*)&lB[buf][r1 * 40 + s1c * 8], 0, 0);
#else
    *(v8h*)&lA[buf][r0 * 40 + s0c * 8] = *(const v8h*)(A + (size_t)(bm + r0) * DD + k + s0c * 8);
    *(v8h*)&lA[buf][r1 * 40 + s1c * 8] = *(const v8h*)(A + (size_t)(bm + r1) * DD + k + s1c * 8);
    *(v8h*)&lB[buf][r0 * 40 + s0c * 8] = *(const v8h*)(Bw + (size_t)(bn + r0) * DD + k + s0c * 8);
    *(v8h*)&lB[buf][r1 * 40 + s1c * 8] = *(const v8h*)(Bw + (size_t)(bn + r1) * DD + k + s1c * 8);
#endif
  };

  v8f acc[4] = {};
  stage(0, 0);
#pragma unroll 1
  for (int i = 0; i < NSTEP; ++i) {
    int buf = i & 1;
    if (i + 1 < NSTEP) {
      stage(buf ^ 1, (i + 1) * 32); // prefetch next panel while current computes
#if HAVE_TDM
      if (wave == 0) WAIT_TENSOR(2); // 2 just issued may fly; current panel done
#elif HAVE_ASYNC
      WAIT_ASYNC(4);
#endif
    } else {
#if HAVE_TDM
      if (wave == 0) WAIT_TENSOR(0);
#elif HAVE_ASYNC
      WAIT_ASYNC(0);
#endif
    }
    __syncthreads();
    const _Float16* pa = &lA[buf][(wave * 16 + l16) * 40 + hf * 8];
    v16h af = ld16(pa, pa + 16);
#pragma unroll
    for (int tt = 0; tt < 4; ++tt) {
      const _Float16* pb = &lB[buf][(tt * 16 + l16) * 40 + hf * 8];
      v16h bf = ld16(pb, pb + 16);
      acc[tt] = wmma_f16(af, bf, acc[tt]);
    }
    __syncthreads();
  }

#pragma unroll
  for (int tt = 0; tt < 4; ++tt) {
    int col = bn + tt * 16 + l16;
    float bs = bias ? bias[col] : 0.f;
#pragma unroll
    for (int r = 0; r < 8; ++r) {
      int m = bm + wave * 16 + hf * 8 + r;
      float vv = acc[tt][r] + bs;
      if (mode == 0) {
        size_t idx = (size_t)m * DD + col;
        if (resid) vv += resid[idx];
        outf[idx] = vv;
      } else {
        int b = m >> 10, s = m & 1023;
        int h = col >> 6, d = col & 63;
        if (mode == 1)      outh[((((size_t)b * HHZ + h) * SSZ + s) << 6) + d] = (_Float16)vv;
        else if (mode == 2) outh[((((size_t)b * HHZ + h) << 6) + d) * SSZ + s] = (_Float16)vv;
        else                outf[((((size_t)b * HHZ + h) * SSZ + s) << 6) + d] = vv;
      }
    }
  }
}

// Flash attention: one wave32 per (b, h, 16-row q block). Streams K/V in 32-col
// chunks; logits via 2 WMMA / 16x16 tile, online softmax, P transposed through
// LDS, P*V via 4 WMMA. Epilogue fuses /rowsum and elementwise R gate; writes
// f16 [B*S, H*DH] ready for the Wo GEMM.
__global__ void attn_k(const _Float16* __restrict__ Qp, const _Float16* __restrict__ Kp,
                       const _Float16* __restrict__ Vt, const float* __restrict__ Rp,
                       const int* __restrict__ mask, _Float16* __restrict__ nv) {
  int qb = blockIdx.x, h = blockIdx.y, b = blockIdx.z;
  int lane = threadIdx.x, l16 = lane & 15, hf = lane >> 4;
  int q0 = qb * 16;
  size_t bh = (size_t)b * HHZ + h;

  const _Float16* qrow = Qp + (bh * SSZ + q0 + l16) * DQ;
  v16h qf0 = ld16(qrow + hf * 8,      qrow + hf * 8 + 16);
  v16h qf1 = ld16(qrow + 32 + hf * 8, qrow + 32 + hf * 8 + 16);

  v8f o0 = {}, o1 = {}, o2 = {}, o3 = {};
  float mmax[8], msum[8];
#pragma unroll
  for (int r = 0; r < 8; ++r) { mmax[r] = -3.0e38f; msum[r] = 0.f; }

  __shared__ __align__(16) _Float16 pt[16 * 32];
  const float scale = 0.125f; // 1/sqrt(64)
  const _Float16* vbase = Vt + bh * DQ * SSZ;

  for (int j = 0; j < SSZ; j += 32) {
    const _Float16* kr0 = Kp + (bh * SSZ + j + l16) * DQ;
    const _Float16* kr1 = kr0 + 16 * DQ;
    v8f s0 = {}, s1 = {};
    v16h kb;
    kb = ld16(kr0 + hf * 8,      kr0 + hf * 8 + 16);      s0 = wmma_f16(qf0, kb, s0);
    kb = ld16(kr0 + 32 + hf * 8, kr0 + 32 + hf * 8 + 16); s0 = wmma_f16(qf1, kb, s0);
    kb = ld16(kr1 + hf * 8,      kr1 + hf * 8 + 16);      s1 = wmma_f16(qf0, kb, s1);
    kb = ld16(kr1 + 32 + hf * 8, kr1 + 32 + hf * 8 + 16); s1 = wmma_f16(qf1, kb, s1);

    int mv0 = mask[(size_t)b * SSZ + j + l16];
    int mv1 = mask[(size_t)b * SSZ + j + 16 + l16];
    float cm[8], fac[8], ps[8];
#pragma unroll
    for (int r = 0; r < 8; ++r) {
      s0[r] = mv0 ? s0[r] * scale : -1e10f;
      s1[r] = mv1 ? s1[r] * scale : -1e10f;
      cm[r] = fmaxf(s0[r], s1[r]);
    }
#pragma unroll
    for (int off = 8; off >= 1; off >>= 1)
#pragma unroll
      for (int r = 0; r < 8; ++r) cm[r] = fmaxf(cm[r], __shfl_xor(cm[r], off, 32));
#pragma unroll
    for (int r = 0; r < 8; ++r) {
      float nm = fmaxf(mmax[r], cm[r]);
      fac[r] = __expf(mmax[r] - nm);
      mmax[r] = nm;
      s0[r] = __expf(s0[r] - nm);
      s1[r] = __expf(s1[r] - nm);
      ps[r] = s0[r] + s1[r];
    }
#pragma unroll
    for (int off = 8; off >= 1; off >>= 1)
#pragma unroll
      for (int r = 0; r < 8; ++r) ps[r] += __shfl_xor(ps[r], off, 32);
#pragma unroll
    for (int r = 0; r < 8; ++r) {
      msum[r] = msum[r] * fac[r] + ps[r];
      o0[r] *= fac[r]; o1[r] *= fac[r]; o2[r] *= fac[r]; o3[r] *= fac[r];
      int m = r + hf * 8;
      pt[m * 32 + l16]      = (_Float16)s0[r];
      pt[m * 32 + 16 + l16] = (_Float16)s1[r];
    }
    // same-wave LDS store->load: DS pipe is in-order; force completion + compiler barrier
    asm volatile("s_wait_dscnt 0" ::: "memory");
    v16h pf = ld16(&pt[l16 * 32 + hf * 8], &pt[l16 * 32 + hf * 8 + 16]);

    const _Float16* v0 = vbase + (size_t)(0 * 16 + l16) * SSZ + j + hf * 8;
    const _Float16* v1 = vbase + (size_t)(1 * 16 + l16) * SSZ + j + hf * 8;
    const _Float16* v2 = vbase + (size_t)(2 * 16 + l16) * SSZ + j + hf * 8;
    const _Float16* v3 = vbase + (size_t)(3 * 16 + l16) * SSZ + j + hf * 8;
    o0 = wmma_f16(pf, ld16(v0, v0 + 16), o0);
    o1 = wmma_f16(pf, ld16(v1, v1 + 16), o1);
    o2 = wmma_f16(pf, ld16(v2, v2 + 16), o2);
    o3 = wmma_f16(pf, ld16(v3, v3 + 16), o3);
  }

  float inv[8];
#pragma unroll
  for (int r = 0; r < 8; ++r) inv[r] = 1.f / msum[r];
  const float* rr = Rp + (bh * SSZ + q0) * DQ;
#pragma unroll
  for (int r = 0; r < 8; ++r) {
    int m = r + hf * 8;
    size_t orow = ((size_t)b * SSZ + q0 + m) * DD + h * DQ;
    const float* rm = rr + (size_t)m * DQ;
    nv[orow + 0 * 16 + l16] = (_Float16)(o0[r] * inv[r] * rm[0 * 16 + l16]);
    nv[orow + 1 * 16 + l16] = (_Float16)(o1[r] * inv[r] * rm[1 * 16 + l16]);
    nv[orow + 2 * 16 + l16] = (_Float16)(o2[r] * inv[r] * rm[2 * 16 + l16]);
    nv[orow + 3 * 16 + l16] = (_Float16)(o3[r] * inv[r] * rm[3 * 16 + l16]);
  }
}

// Cox-de Boor cubic B-spline on uniform knots linspace(-1,1,8); first two basis fns.
static __device__ __forceinline__ void bspline01(float x, float* b0, float* b1) {
  const float t0 = -1.f, hk = 2.f / 7.f;
  float Bv[7];
#pragma unroll
  for (int k = 0; k < 7; ++k) {
    float tk = t0 + k * hk, tk1 = t0 + (k + 1) * hk;
    Bv[k] = (tk <= x && x < tk1) ? 1.f : 0.f;
  }
#pragma unroll
  for (int m = 1; m <= 3; ++m) {
#pragma unroll
    for (int k = 0; k + m < 7; ++k) {
      float tk   = t0 + k * hk;
      float tkm  = t0 + (k + m) * hk;
      float tk1  = t0 + (k + 1) * hk;
      float tkm1 = t0 + (k + m + 1) * hk;
      Bv[k] = (x - tk) / (tkm - tk) * Bv[k] + (tkm1 - x) / (tkm1 - tk1) * Bv[k + 1];
    }
  }
  *b0 = Bv[0]; *b1 = Bv[1];
}

// Fused LN2 + tanh + bspline + inner_c contraction -> f16 GEMM input.
__global__ void kan_k(const float* __restrict__ x, const float* __restrict__ w,
                      const float* __restrict__ b, const float* __restrict__ ic,
                      _Float16* __restrict__ out16) {
  int row = blockIdx.x, t = threadIdx.x;
  const float* xr = x + (size_t)row * DD;
  float v[4]; float s = 0.f, q = 0.f;
#pragma unroll
  for (int i = 0; i < 4; ++i) { v[i] = xr[t + i * 128]; s += v[i]; q += v[i] * v[i]; }
  __shared__ float rs[128], rq[128];
  rs[t] = s; rq[t] = q; __syncthreads();
  for (int off = 64; off; off >>= 1) {
    if (t < off) { rs[t] += rs[t + off]; rq[t] += rq[t + off]; }
    __syncthreads();
  }
  float mu = rs[0] * (1.f / DD);
  float var = rq[0] * (1.f / DD) - mu * mu;
  float rstd = rsqrtf(var + 1e-5f);
#pragma unroll
  for (int i = 0; i < 4; ++i) {
    int c = t + i * 128;
    float y  = (v[i] - mu) * rstd * w[c] + b[c];
    float xt = tanhf(y);
    float b0, b1; bspline01(xt, &b0, &b1);
    float inner = b0 * ic[c * 5 + 0] + b1 * ic[c * 5 + 1];
    out16[(size_t)row * DD + c] = (_Float16)inner;
  }
}

// ---------------------------------------------------------------- host
extern "C" void kernel_launch(void* const* d_in, const int* in_sizes, int n_in,
                              void* d_out, int out_size, void* d_ws, size_t ws_size,
                              hipStream_t stream) {
  (void)in_sizes; (void)n_in; (void)out_size; (void)ws_size;
  const float* src  = (const float*)d_in[0];
  const float* ln1w = (const float*)d_in[1];  const float* ln1b = (const float*)d_in[2];
  const float* ln2w = (const float*)d_in[3];  const float* ln2b = (const float*)d_in[4];
  const float* ln3w = (const float*)d_in[5];  const float* ln3b = (const float*)d_in[6];
  const float* Wq_w = (const float*)d_in[7];  const float* Wq_b = (const float*)d_in[8];
  const float* Wk_w = (const float*)d_in[9];  const float* Wk_b = (const float*)d_in[10];
  const float* Wv_w = (const float*)d_in[11]; const float* Wv_b = (const float*)d_in[12];
  const float* Wr_w = (const float*)d_in[13]; const float* Wr_b = (const float*)d_in[14];
  const float* Wo_w = (const float*)d_in[15]; const float* Wo_b = (const float*)d_in[16];
  const float* innc = (const float*)d_in[17]; const float* outc = (const float*)d_in[18];
  const int*   msk  = (const int*)d_in[19];
  float* cur = (float*)d_out; // residual stream lives in d_out

  size_t off = 0;
  auto alloc = [&](size_t bytes) -> void* {
    void* p = (char*)d_ws + off;
    off += (bytes + 255) & ~(size_t)255;
    return p;
  };
  const size_t WN = (size_t)LLZ * 512 * 512; // elems per weight tensor (all layers)
  _Float16* wq16  = (_Float16*)alloc(WN * 2);
  _Float16* wk16  = (_Float16*)alloc(WN * 2);
  _Float16* wv16  = (_Float16*)alloc(WN * 2);
  _Float16* wr16  = (_Float16*)alloc(WN * 2);
  _Float16* wo16  = (_Float16*)alloc(WN * 2);
  _Float16* woc16 = (_Float16*)alloc(WN * 2);
  _Float16* z16 = (_Float16*)alloc((size_t)ROWS * DD * 2);
  _Float16* x16 = (_Float16*)alloc((size_t)ROWS * DD * 2);
  _Float16* Qp  = (_Float16*)alloc((size_t)BBZ * HHZ * SSZ * DQ * 2);
  _Float16* Kp  = (_Float16*)alloc((size_t)BBZ * HHZ * SSZ * DQ * 2);
  _Float16* Vt  = (_Float16*)alloc((size_t)BBZ * HHZ * SSZ * DQ * 2);
  float*    Rp  = (float*)alloc((size_t)BBZ * HHZ * SSZ * DQ * 4);

  (void)hipMemcpyAsync(cur, src, (size_t)ROWS * DD * 4, hipMemcpyDeviceToDevice, stream);

  int wb = (int)((WN + 255) / 256);
  f32_to_f16_k<<<wb, 256, 0, stream>>>(Wq_w, wq16, (int)WN);
  f32_to_f16_k<<<wb, 256, 0, stream>>>(Wk_w, wk16, (int)WN);
  f32_to_f16_k<<<wb, 256, 0, stream>>>(Wv_w, wv16, (int)WN);
  f32_to_f16_k<<<wb, 256, 0, stream>>>(Wr_w, wr16, (int)WN);
  f32_to_f16_k<<<wb, 256, 0, stream>>>(Wo_w, wo16, (int)WN);
  f32_to_f16_k<<<wb, 256, 0, stream>>>(outc, woc16, (int)WN);

  dim3 gg(DD / 64, ROWS / 64);   // 8 x 64 blocks of 64x64
  dim3 ag(SSZ / 16, HHZ, BBZ);   // 64 x 8 x 4

  for (int l = 0; l < LLZ; ++l) {
    size_t wo = (size_t)l * 512 * 512;
    ln_k<<<ROWS, 128, 0, stream>>>(cur, ln1w + l * DD, ln1b + l * DD, z16, nullptr);

    gemm_k<<<gg, 128, 0, stream>>>(z16, wq16 + wo, Wq_b + l * DD, nullptr, nullptr, Qp, 1);
    gemm_k<<<gg, 128, 0, stream>>>(z16, wk16 + wo, Wk_b + l * DD, nullptr, nullptr, Kp, 1);
    gemm_k<<<gg, 128, 0, stream>>>(z16, wv16 + wo, Wv_b + l * DD, nullptr, nullptr, Vt, 2);
    gemm_k<<<gg, 128, 0, stream>>>(z16, wr16 + wo, Wr_b + l * DD, nullptr, Rp, nullptr, 3);

    attn_k<<<ag, 32, 0, stream>>>(Qp, Kp, Vt, Rp, msk, x16);

    gemm_k<<<gg, 128, 0, stream>>>(x16, wo16 + wo, Wo_b + l * DD, cur, cur, nullptr, 0);

    kan_k<<<ROWS, 128, 0, stream>>>(cur, ln2w + l * DD, ln2b + l * DD, innc + (size_t)l * DD * 5, x16);
    gemm_k<<<gg, 128, 0, stream>>>(x16, woc16 + wo, nullptr, cur, cur, nullptr, 0);

    ln_k<<<ROWS, 128, 0, stream>>>(cur, ln3w + l * DD, ln3b + l * DD, nullptr, cur);
  }
}
